// AutoRegressiveGaussianDistribution_84086869721323
// MI455X (gfx1250) — matrix operand order repaired and checked by reading, not compiled, not run on hardware
//
#include <hip/hip_runtime.h>

typedef __attribute__((ext_vector_type(16))) __bf16 v16bf;
typedef __attribute__((ext_vector_type(8)))  float  v8f;

#define D_  64
#define C_  128
#define H_  512
#define RT  32      // batch rows per workgroup

struct __align__(32) BFrag { uint4 lo, hi; };

// Degrees sorted ascending: degree d in 1..63, count 9 for d<=8 else 8.
__device__ __forceinline__ int degOf(int p)  { return (p < 72) ? (p / 9 + 1) : ((p - 72) / 8 + 9); }
__device__ __forceinline__ int origOf(int p) {
  int d, r;
  if (p < 72) { d = p / 9 + 1; r = p % 9; }
  else        { d = (p - 72) / 8 + 9; r = (p - 72) % 8; }
  return (d - 1) + 63 * r;            // original hidden index with degree d
}
__device__ __forceinline__ int cumOf(int d)  { return 8 * d + (d < 8 ? d : 8); } // #units with degree <= d

__device__ __forceinline__ unsigned short f2bf(float x) {
  __bf16 h = (__bf16)x;
  return __builtin_bit_cast(unsigned short, h);
}
__device__ __forceinline__ unsigned int pack2(float lo, float hi) {
  return ((unsigned int)f2bf(hi) << 16) | (unsigned int)f2bf(lo);
}
__device__ __forceinline__ float softplus_f(float x) {
  return (x > 20.f) ? x : log1pf(__expf(x));
}
// bf16 pair (packed in a dword) -> f32 via bit ops (bf16 is the top half of f32)
__device__ __forceinline__ float bflo(unsigned int u) { return __builtin_bit_cast(float, u << 16); }
__device__ __forceinline__ float bfhi(unsigned int u) { return __builtin_bit_cast(float, u & 0xffff0000u); }
__device__ __forceinline__ float dot8(uint4 a, uint4 b) {
  float s;
  s  = bflo(a.x) * bflo(b.x);  s = fmaf(bfhi(a.x), bfhi(b.x), s);
  s  = fmaf(bflo(a.y), bflo(b.y), s);  s = fmaf(bfhi(a.y), bfhi(b.y), s);
  s  = fmaf(bflo(a.z), bflo(b.z), s);  s = fmaf(bfhi(a.z), bfhi(b.z), s);
  s  = fmaf(bflo(a.w), bflo(b.w), s);  s = fmaf(bfhi(a.w), bfhi(b.w), s);
  return s;
}

// ---------------- prep kernels: mask + degree-sort + bf16 + fragment-swizzle ----------------
// B-fragment blob for a K32xN16 tile: lane l (n = l&15, kbase = (l>>4)*16) holds
// 8 dwords, dword j = pack(W[k0+2j][n], W[k0+2j+1][n]).  1 KB per tile.

__global__ void prep_w1(const float* __restrict__ W1, unsigned int* __restrict__ W1f) {
  int gid = blockIdx.x * blockDim.x + threadIdx.x;
  if (gid >= 6 * 32 * 32 * 8) return;
  int j  = gid & 7;
  int ln = (gid >> 3) & 31;
  int nt = (gid >> 8) & 31;
  int kt = gid >> 13;                       // 0..5  (K = 192)
  int n  = ln & 15, kbase = (ln >> 4) * 16;
  int ps = nt * 16 + n;                     // sorted hidden index
  int h  = origOf(ps);
  int dh = degOf(ps);
  int k0 = kt * 32 + kbase + 2 * j;
  float v0 = W1[(size_t)k0 * H_ + h];
  float v1 = W1[(size_t)(k0 + 1) * H_ + h];
  if (k0     < D_ && dh < k0 + 1) v0 = 0.f; // M1: z row d connects iff deg >= d+1
  if (k0 + 1 < D_ && dh < k0 + 2) v1 = 0.f; // ctx rows (k>=64) unmasked
  W1f[(size_t)(kt * 32 + nt) * 256 + ln * 8 + j] = pack2(v0, v1);
}

__global__ void prep_w2(const float* __restrict__ W2, unsigned int* __restrict__ W2f) {
  int gid = blockIdx.x * blockDim.x + threadIdx.x;
  if (gid >= 16 * 32 * 32 * 8) return;
  int j  = gid & 7;
  int ln = (gid >> 3) & 31;
  int nt = (gid >> 8) & 31;
  int kt = gid >> 13;                       // 0..15 (K = 512)
  int n  = ln & 15, kbase = (ln >> 4) * 16;
  int ps = nt * 16 + n;
  int h  = origOf(ps);
  int dh = degOf(ps);
  int k0 = kt * 32 + kbase + 2 * j;         // sorted input-unit index
  float v0 = (dh >= degOf(k0))     ? W2[(size_t)origOf(k0) * H_ + h]     : 0.f;
  float v1 = (dh >= degOf(k0 + 1)) ? W2[(size_t)origOf(k0 + 1) * H_ + h] : 0.f;
  W2f[(size_t)(kt * 32 + nt) * 256 + ln * 8 + j] = pack2(v0, v1);
}

__global__ void prep_w3t(const float* __restrict__ W3, __bf16* __restrict__ W3T) {
  int gid = blockIdx.x * blockDim.x + threadIdx.x;
  if (gid >= 128 * H_) return;
  int j = gid >> 9;          // output column 0..127
  int p = gid & 511;         // sorted hidden index
  int d = degOf(p);
  float v = (((j & 63) + 1) > d) ? W3[(size_t)origOf(p) * 128 + j] : 0.f;
  W3T[(size_t)j * H_ + p] = (__bf16)v;
}

__global__ void prep_bias(const float* __restrict__ b1, const float* __restrict__ b2,
                          float* __restrict__ b1s, float* __restrict__ b2s) {
  int gid = blockIdx.x * blockDim.x + threadIdx.x;
  if (gid >= H_) return;
  int h = origOf(gid);
  b1s[gid] = b1[h];
  b2s[gid] = b2[h];
}

// ---------------- fragment loads ----------------
// 16-bit A (16x32) layout: lane l: m = l&15, klo = (l>>4)*16; v0..7 = K klo..klo+15 (contiguous, matches B blobs).
__device__ __forceinline__ v16bf loadA(const __bf16* base, int rowStride, int mRow0, int kCol0, int lane) {
  int m   = mRow0 + (lane & 15);
  int klo = (lane >> 4) * 16;
  const __bf16* p = base + (size_t)m * rowStride + kCol0 + klo;
  BFrag f;
  f.lo = *reinterpret_cast<const uint4*>(p);
  f.hi = *reinterpret_cast<const uint4*>(p + 8);
  return __builtin_bit_cast(v16bf, f);
}
__device__ __forceinline__ v16bf loadB(const unsigned int* Wf, int tileIdx, int lane) {
  const unsigned int* b = Wf + (size_t)tileIdx * 256 + lane * 8;
  BFrag f;
  f.lo = *reinterpret_cast<const uint4*>(b);
  f.hi = *reinterpret_cast<const uint4*>(b + 4);
  return __builtin_bit_cast(v16bf, f);
}

// ---------------- main: one WG owns 32 batch rows for all 64 autoregressive steps ----------------
__global__ __launch_bounds__(256)
void made_autoregressive_sample(const float* __restrict__ ctx,   // (B,128)
                                const float* __restrict__ eps,   // (B,64)
                                const unsigned int* __restrict__ W1f,
                                const unsigned int* __restrict__ W2f,
                                const __bf16* __restrict__ W3T,  // (128,512) masked, sorted
                                const float* __restrict__ b1s,
                                const float* __restrict__ b2s,
                                const float* __restrict__ b3,    // (128,)
                                float* __restrict__ out)         // (B,64)
{
  __shared__ __align__(16) __bf16 xa[RT][D_ + C_];   // GEMM1 A: [z | ctx], bf16
  __shared__ __align__(16) __bf16 h1s[RT][H_];
  __shared__ __align__(16) __bf16 h2s[RT][H_];
  __shared__ float zf[RT][D_];
  __shared__ float dotp[RT][2][4];
  __shared__ float mres[RT], pres[RT];

  const int tid  = threadIdx.x;
  const int lane = tid & 31;
  const int wave = tid >> 5;
  const int row0 = blockIdx.x * RT;

  // init: z=0, ctx->bf16, h1/h2 zeroed (so stale columns are finite zeros under masked weights)
  for (int idx = tid; idx < RT * (D_ + C_); idx += 256) {
    int r = idx / (D_ + C_), c = idx % (D_ + C_);
    xa[r][c] = (c < D_) ? (__bf16)0.f : (__bf16)ctx[(size_t)(row0 + r) * C_ + (c - D_)];
  }
  for (int idx = tid; idx < RT * D_; idx += 256) zf[idx >> 6][idx & 63] = 0.f;
  for (int idx = tid; idx < RT * H_; idx += 256) { h1s[idx >> 9][idx & 511] = (__bf16)0.f;
                                                  h2s[idx >> 9][idx & 511] = (__bf16)0.f; }
  __syncthreads();

  // step 0: output col 0 depends only on biases
  if (tid < RT) {
    float z = b3[0] + softplus_f(b3[D_]) * eps[(size_t)(row0 + tid) * D_ + 0];
    zf[tid][0] = z;
    xa[tid][0] = (__bf16)z;
  }
  __syncthreads();

  for (int i = 1; i < D_; ++i) {
    const int Nact = cumOf(i);              // hidden units with degree <= i
    const int Nt   = (Nact + 15) >> 4;      // active 16-col tiles
    const int kz   = (i + 31) >> 5;         // nonzero-z K32 tiles in GEMM1

    // ---- GEMM1: x(32x192) @ W1m -> relu -> h1 (first Nt tiles) ----
    for (int t = wave; t < 2 * Nt; t += 8) {
      int mt = t & 1, nt = t >> 1;
      v8f acc = {0.f, 0.f, 0.f, 0.f, 0.f, 0.f, 0.f, 0.f};
      for (int kt = 0; kt < kz; ++kt) {     // z prefix (<= 2 tiles)
        v16bf a = loadA(&xa[0][0], D_ + C_, mt * 16, kt * 32, lane);
        v16bf b = loadB(W1f, kt * 32 + nt, lane);
        acc = __builtin_amdgcn_wmma_f32_16x16x32_bf16(false, a, false, b, (short)0, acc, false, false);
      }
#pragma unroll
      for (int kt = 2; kt < 6; ++kt) {      // context (always dense)
        v16bf a = loadA(&xa[0][0], D_ + C_, mt * 16, kt * 32, lane);
        v16bf b = loadB(W1f, kt * 32 + nt, lane);
        acc = __builtin_amdgcn_wmma_f32_16x16x32_bf16(false, a, false, b, (short)0, acc, false, false);
      }
      int col = nt * 16 + (lane & 15);
      int rhi = (lane >> 4) * 8;
      float bb = b1s[col];
#pragma unroll
      for (int r = 0; r < 8; ++r) {
        float v = acc[r] + bb;
        h1s[mt * 16 + rhi + r][col] = (__bf16)(v > 0.f ? v : 0.f);
      }
    }
    __syncthreads();

    // ---- GEMM2: h1(32x512) @ W2m -> relu -> h2 ----
    // triangular K bound; manually unrolled x4 with 4 independent accumulator
    // chains so the 16 fragment loads clause together and WMMAs overlap the
    // waits of later tiles (the compiler re-rolls rotated pipelines otherwise)
    for (int t = wave; t < 2 * Nt; t += 8) {
      int mt = t & 1, nt = t >> 1;
      int dhi    = degOf(nt * 16 + 15 < H_ - 1 ? nt * 16 + 15 : H_ - 1);
      int kunits = cumOf(dhi); if (kunits > Nact) kunits = Nact;
      int kt2    = (kunits + 31) >> 5; if (kt2 < 1) kt2 = 1;
      v8f acc0 = {0.f, 0.f, 0.f, 0.f, 0.f, 0.f, 0.f, 0.f};
      v8f acc1 = acc0, acc2 = acc0, acc3 = acc0;
      const __bf16* A = &h1s[0][0];
      int kt = 0;
      for (; kt + 4 <= kt2; kt += 4) {
        v16bf a0 = loadA(A, H_, mt * 16, (kt + 0) * 32, lane);
        v16bf b0 = loadB(W2f, (kt + 0) * 32 + nt, lane);
        v16bf a1 = loadA(A, H_, mt * 16, (kt + 1) * 32, lane);
        v16bf b1 = loadB(W2f, (kt + 1) * 32 + nt, lane);
        v16bf a2 = loadA(A, H_, mt * 16, (kt + 2) * 32, lane);
        v16bf b2 = loadB(W2f, (kt + 2) * 32 + nt, lane);
        v16bf a3 = loadA(A, H_, mt * 16, (kt + 3) * 32, lane);
        v16bf b3f = loadB(W2f, (kt + 3) * 32 + nt, lane);
        acc0 = __builtin_amdgcn_wmma_f32_16x16x32_bf16(false, a0, false, b0,  (short)0, acc0, false, false);
        acc1 = __builtin_amdgcn_wmma_f32_16x16x32_bf16(false, a1, false, b1,  (short)0, acc1, false, false);
        acc2 = __builtin_amdgcn_wmma_f32_16x16x32_bf16(false, a2, false, b2,  (short)0, acc2, false, false);
        acc3 = __builtin_amdgcn_wmma_f32_16x16x32_bf16(false, a3, false, b3f, (short)0, acc3, false, false);
      }
      for (; kt < kt2; ++kt) {              // tail (<= 3 tiles)
        v16bf a = loadA(A, H_, mt * 16, kt * 32, lane);
        v16bf b = loadB(W2f, kt * 32 + nt, lane);
        acc0 = __builtin_amdgcn_wmma_f32_16x16x32_bf16(false, a, false, b, (short)0, acc0, false, false);
      }
      int col = nt * 16 + (lane & 15);
      int rhi = (lane >> 4) * 8;
      float bb = b2s[col];
#pragma unroll
      for (int r = 0; r < 8; ++r) {
        float v = (acc0[r] + acc1[r]) + (acc2[r] + acc3[r]) + bb;
        h2s[mt * 16 + rhi + r][col] = (__bf16)(v > 0.f ? v : 0.f);
      }
    }
    __syncthreads();

    // ---- GEMM3: only columns i and 64+i -> vectorized VALU dots over active units ----
    {
      int r   = tid & 31;
      int rem = tid >> 5;
      int wch = rem & 1;                    // 0 = mean, 1 = prescale
      int seg = rem >> 1;                   // 4 K-segments of 128
      int kend = Nt * 16;                   // multiple of 16
      const __bf16* w3 = W3T + (size_t)(wch * D_ + i) * H_;
      int pb = seg * 128;
      int pe = pb + 128; if (pe > kend) pe = kend;
      float s = 0.f;
      for (int p = pb; p < pe; p += 8) {    // b128 LDS + b128 global per 8 elements
        uint4 hv = *reinterpret_cast<const uint4*>(&h2s[r][p]);
        uint4 wv = *reinterpret_cast<const uint4*>(w3 + p);
        s += dot8(hv, wv);
      }
      dotp[r][wch][seg] = s;
    }
    __syncthreads();
    if (tid < 64) {
      int r = tid & 31, wch = tid >> 5;
      float s = dotp[r][wch][0] + dotp[r][wch][1] + dotp[r][wch][2] + dotp[r][wch][3]
              + b3[wch * D_ + i];
      if (wch == 0) mres[r] = s; else pres[r] = s;
    }
    __syncthreads();
    if (tid < RT) {
      float z = mres[tid] + softplus_f(pres[tid]) * eps[(size_t)(row0 + tid) * D_ + i];
      zf[tid][i] = z;
      xa[tid][i] = (__bf16)z;
    }
    __syncthreads();
  }

  for (int idx = tid; idx < RT * D_; idx += 256) {
    int r = idx >> 6, c = idx & 63;
    out[(size_t)(row0 + r) * D_ + c] = zf[r][c];
  }
}

// ---------------- launch ----------------
extern "C" void kernel_launch(void* const* d_in, const int* in_sizes, int n_in,
                              void* d_out, int out_size, void* d_ws, size_t ws_size,
                              hipStream_t stream) {
  (void)n_in; (void)out_size; (void)ws_size;
  const float* ctx = (const float*)d_in[0];   // (B,128)
  const float* eps = (const float*)d_in[1];   // (B,64)
  const float* W1  = (const float*)d_in[2];   // (192,512)
  const float* b1  = (const float*)d_in[3];   // (512,)
  const float* W2  = (const float*)d_in[4];   // (512,512)
  const float* b2  = (const float*)d_in[5];   // (512,)
  const float* W3  = (const float*)d_in[6];   // (512,128)
  const float* b3  = (const float*)d_in[7];   // (128,)
  float* out = (float*)d_out;

  const int B = in_sizes[1] / D_;

  // workspace layout (bytes): W1f 192K | W2f 512K | W3T 128K | b1s 2K | b2s 2K
  unsigned int* W1f = (unsigned int*)d_ws;
  unsigned int* W2f = W1f + (size_t)6 * 32 * 256;
  __bf16* W3T = (__bf16*)(W2f + (size_t)16 * 32 * 256);
  float* b1s = (float*)(W3T + (size_t)128 * H_);
  float* b2s = b1s + H_;

  prep_w1 <<<dim3((6 * 32 * 32 * 8 + 255) / 256), dim3(256), 0, stream>>>(W1, W1f);
  prep_w2 <<<dim3((16 * 32 * 32 * 8 + 255) / 256), dim3(256), 0, stream>>>(W2, W2f);
  prep_w3t<<<dim3((128 * H_ + 255) / 256), dim3(256), 0, stream>>>(W3, W3T);
  prep_bias<<<dim3((H_ + 255) / 256), dim3(256), 0, stream>>>(b1, b2, b1s, b2s);

  made_autoregressive_sample<<<dim3(B / RT), dim3(256), 0, stream>>>(
      ctx, eps, W1f, W2f, W3T, b1s, b2s, b3, out);
}
